// TransformerUnit_73194832658888
// MI455X (gfx1250) — compile-verified, matrix operand order checked by reading, not run on hardware
//
#include <hip/hip_runtime.h>
#include <hip/hip_bf16.h>
#include <stdint.h>

// ---------------------------------------------------------------------------
// Problem dims (compile-time constants from the reference)
// ---------------------------------------------------------------------------
constexpr int  B_   = 4;
constexpr int  S_   = 2048;
constexpr int  D_   = 256;
constexpr int  H_   = 8;
constexpr int  FD_  = 512;             // F*D
constexpr long M_   = (long)B_ * S_;   // 8192 flattened rows
constexpr long HD_  = (long)H_ * D_;   // 2048 concat width

typedef __attribute__((ext_vector_type(16))) __bf16 bf16x16;
typedef __attribute__((ext_vector_type(8)))  __bf16 bf16x8;
typedef __attribute__((ext_vector_type(8)))  float  f32x8;
typedef __attribute__((ext_vector_type(4)))  unsigned int u32x4;
typedef __attribute__((ext_vector_type(8)))  int i32x8;
typedef __attribute__((ext_vector_type(4)))  int i32x4;

#if defined(__has_builtin)
#if __has_builtin(__builtin_amdgcn_tensor_load_to_lds)
#define HAVE_TDM 1
#else
#define HAVE_TDM 0
#endif
#else
#define HAVE_TDM 0
#endif

// ---------------------------------------------------------------------------
// WMMA helpers.  CDNA5 bf16 A/B fragment layout (16x32 / 32x16):
//   lanes 0..15  : row=lane,    K = [0..7]  and [16..23]
//   lanes 16..31 : row=lane-16, K = [8..15] and [24..31]
// With contiguous-K data a fragment is two 16-byte loads + a shuffle.
// ---------------------------------------------------------------------------
__device__ __forceinline__ bf16x16 frag_ld_g(const __bf16* p) {
  bf16x8 lo = *(const bf16x8*)(p);
  bf16x8 hi = *(const bf16x8*)(p + 16);
  return __builtin_shufflevector(lo, hi, 0,1,2,3,4,5,6,7,8,9,10,11,12,13,14,15);
}

__device__ __forceinline__ f32x8 wmma_bf16(bf16x16 a, bf16x16 b, f32x8 c) {
  return __builtin_amdgcn_wmma_f32_16x16x32_bf16(
      /*neg_a=*/false, a, /*neg_b=*/false, b,
      /*c_mod=*/(short)0, c, /*reuse_a=*/false, /*reuse_b=*/false);
}

#if HAVE_TDM
// ---------------------------------------------------------------------------
// TDM: DMA a 64(row) x 32(k) bf16 tile (row stride ldbt elems) from global
// into LDS at byte offset ldsOff, inserting 16B of padding after every 64B
// row (pad_interval=16 DWORDs -> code 3, pad_amount=4 DWORDs -> code 3),
// i.e. an 80-byte LDS row pitch for bank-conflict-free b128 fragment reads.
// Toolchain probe: 6-arg builtin (g0:u32x4, g1:i32x8, g2:i32x4, g3:i32x4,
// g4:i32x8, cpol:i32).
// ---------------------------------------------------------------------------
__device__ __forceinline__ void tdm_load_tile(const __bf16* gsrc,
                                              unsigned ldsOff, long ldbt) {
  unsigned long long ga = (unsigned long long)(uintptr_t)gsrc;
  u32x4 g0;
  g0[0] = 1u;                                        // count=1, user mode
  g0[1] = ldsOff;                                    // lds_addr (bytes)
  g0[2] = (unsigned)(ga & 0xFFFFFFFFull);            // global_addr[31:0]
  g0[3] = (unsigned)((ga >> 32) & 0x01FFFFFFull)     // global_addr[56:32]
          | 0x80000000u;                             // type=2 ("image")
  i32x8 g1;
  g1[0] = (1 << 16)      // data_size = 2 bytes
        | (1 << 20)      // pad_enable
        | (3 << 22)      // pad_interval: 16 DWORDs
        | (3 << 25);     // pad_amount:   4 DWORDs
  g1[1] = 0;                                   // tensor_dim0 low16 (bits 63:48)
  g1[2] = (1 << 14);                           // tensor_dim0 = 1<<30 (hi bits)
  g1[3] = 16 | (32 << 16);                     // tensor_dim1=1<<20 hi, tile_dim0=32
  g1[4] = 64;                                  // tile_dim1=64, tile_dim2=0
  g1[5] = (int)ldbt;                           // tensor_dim0_stride[31:0]
  g1[6] = 0;                                   // stride hi / dim1_stride lo
  g1[7] = 0;
  i32x4 z4 = {0, 0, 0, 0};
  i32x8 z8 = {0, 0, 0, 0, 0, 0, 0, 0};
  __builtin_amdgcn_tensor_load_to_lds(g0, g1, z4, z4, z8, 0);
}

template <int N>
__device__ __forceinline__ void tdm_wait() {
#if __has_builtin(__builtin_amdgcn_s_wait_tensorcnt)
  __builtin_amdgcn_s_wait_tensorcnt((short)N);
#else
  asm volatile("s_wait_tensorcnt %0" :: "i"(N) : "memory");
#endif
}
#endif  // HAVE_TDM

// ---------------------------------------------------------------------------
// fp32 -> bf16 cast (grid-stride)
// ---------------------------------------------------------------------------
__global__ void cast_bf16_k(const float* __restrict__ in,
                            __bf16* __restrict__ out, long n) {
  long i      = (long)blockIdx.x * blockDim.x + threadIdx.x;
  long stride = (long)gridDim.x * blockDim.x;
  for (; i < n; i += stride) out[i] = (__bf16)in[i];
}

// ---------------------------------------------------------------------------
// fp32 [rows][cols] -> bf16 transposed [cols][rows]   (per-z slice)
// ---------------------------------------------------------------------------
__global__ void transpose_cast_k(const float* __restrict__ in,
                                 __bf16* __restrict__ out,
                                 int rows, int cols,
                                 long inStrideZ, long outStrideZ) {
  __shared__ float tile[16][17];
  const float* inz  = in  + (long)blockIdx.z * inStrideZ;
  __bf16*      outz = out + (long)blockIdx.z * outStrideZ;
  int c = blockIdx.x * 16 + threadIdx.x;
  int r = blockIdx.y * 16 + threadIdx.y;
  if (r < rows && c < cols) tile[threadIdx.y][threadIdx.x] = inz[(long)r * cols + c];
  __syncthreads();
  int oc  = blockIdx.y * 16 + threadIdx.x;   // new col = old row
  int orw = blockIdx.x * 16 + threadIdx.y;   // new row = old col
  if (orw < cols && oc < rows)
    outz[(long)orw * rows + oc] = (__bf16)tile[threadIdx.x][threadIdx.y];
}

// ---------------------------------------------------------------------------
// Generic bf16 WMMA GEMM:  C = act(alpha * A @ B + bias + res)
//   A  : [M][K] bf16 row-major (per-z stride aStrideZ)
//   BT : [N][K] bf16 (B pre-transposed -> contiguous K fragments)
// Block: 256 threads = 8 waves, tile 128(M) x 64(N); wave = 16x64 (4 accums).
// The 64xK B panel is shared by all 8 waves, so it is staged through LDS:
// TDM double-buffered DMA (wave 0 issues, TENSORcnt-ordered) when available,
// else a synchronous cooperative copy.  LDS row pitch 80B (32 k-elems + 8
// pad elems) -> conflict-free ds_load_b128 fragment reads.
// ---------------------------------------------------------------------------
__global__ void __launch_bounds__(256)
gemm_bf16_wmma(const __bf16* __restrict__ A,  long lda,  long aStrideZ,
               const __bf16* __restrict__ BT, long ldbt, long btStrideZ,
               const float* __restrict__ bias,
               const float* __restrict__ res, long ldres,
               float alpha, int relu,
               float*  __restrict__ outF,
               __bf16* __restrict__ outB,  long ldo,  long oStrideZ,
               __bf16* __restrict__ outBT, long ldoT, long oTStrideZ,
               int K) {
  __shared__ __bf16 sB[2][64 * 40];   // 2 x 5120B padded tiles

  const int tid  = threadIdx.x;
  const int lane = tid & 31;
  const int wave = tid >> 5;
  const int z    = blockIdx.z;
  const long m0  = (long)blockIdx.x * 128 + wave * 16;
  const long n0  = (long)blockIdx.y * 64;
  const int  ml  = lane & 15;
  const int  kh  = (lane < 16) ? 0 : 8;   // K half-offset per fragment layout
  const int  rh  = (lane < 16) ? 0 : 8;   // C/D row-half offset

  const __bf16* aRow   = A  + (long)z * aStrideZ + (m0 + ml) * lda + kh;
  const __bf16* Bpanel = BT + (long)z * btStrideZ + n0 * ldbt;  // [64][K]

#if HAVE_TDM
  const unsigned ldsBase = (unsigned)(uintptr_t)(&sB[0][0]);
  if (tid < 32) tdm_load_tile(Bpanel, ldsBase, ldbt);           // prologue
#endif

  f32x8 acc[4] = {};
  const int nsteps = K >> 5;
  for (int s = 0; s < nsteps; ++s) {
    const int k0  = s << 5;
    const int cur = s & 1;
#if HAVE_TDM
    if (tid < 32) {
      if (s + 1 < nsteps) {
        tdm_load_tile(Bpanel + k0 + 32, ldsBase + (unsigned)((cur ^ 1) * 64 * 40 * 2), ldbt);
        tdm_wait<1>();   // previous DMA (in-order) has landed
      } else {
        tdm_wait<0>();
      }
    }
    __syncthreads();
#else
    {   // synchronous cooperative staging: 256 threads x 16B = 4KB tile
      const int n = tid >> 2, c = tid & 3;
      *(bf16x8*)(&sB[cur][n * 40 + c * 8]) =
          *(const bf16x8*)(Bpanel + (long)n * ldbt + k0 + c * 8);
      __syncthreads();
    }
#endif
    bf16x16 afr = frag_ld_g(aRow + k0);
    __builtin_prefetch(aRow + k0 + 128, 0, 1);   // global_prefetch_b8
    const __bf16* bl = &sB[cur][kh];
#pragma unroll
    for (int j = 0; j < 4; ++j) {
      bf16x16 bfr = frag_ld_g(bl + (j * 16 + ml) * 40);
      acc[j] = wmma_bf16(afr, bfr, acc[j]);
    }
    __syncthreads();   // protect buffer before it is overwritten
  }

#pragma unroll
  for (int j = 0; j < 4; ++j) {
    const long n   = n0 + j * 16 + ml;
    const float bv = bias ? bias[n] : 0.0f;
    if (outBT) {
      // transposed store: 8 consecutive M per lane -> one 16B store
      bf16x8 t;
#pragma unroll
      for (int i = 0; i < 8; ++i) {
        long m  = m0 + rh + i;
        float v = acc[j][i] * alpha + bv;
        if (res)  v += res[m * ldres + n];
        if (relu) v = fmaxf(v, 0.0f);
        t[i] = (__bf16)v;
      }
      *(bf16x8*)(outBT + (long)z * oTStrideZ + n * ldoT + (m0 + rh)) = t;
    } else {
#pragma unroll
      for (int i = 0; i < 8; ++i) {
        long m  = m0 + rh + i;
        float v = acc[j][i] * alpha + bv;
        if (res)  v += res[m * ldres + n];
        if (relu) v = fmaxf(v, 0.0f);
        long off = (long)z * oStrideZ + m * ldo + n;
        if (outF) outF[off] = v;
        if (outB) outB[off] = (__bf16)v;
      }
    }
  }
}

// ---------------------------------------------------------------------------
// Attention: one block per (batch, 16-row tile), loop over 8 heads.
//   scores/probs : bf16 LDS [16][2048]  (64 KB)
//   mean accum   : fp32 LDS [16][2048]  (128 KB)   -> 192 KB dynamic LDS
// Stage 1: each wave computes a 16x256 score slab (Q fragments register-
//          resident: 8 x bf16x16 = 64 VGPRs, reused across 16 tiles).
// Stage 2: each wave softmaxes 2 rows (wave32 shfl reductions).
// Stage 3: each wave computes a 16x32 slab of P@V (V pre-transposed).
// attn_mean accumulates in LDS and is written to HBM exactly once.
// ---------------------------------------------------------------------------
__global__ void __launch_bounds__(256)
attn_k(const __bf16* __restrict__ Qb,   // [H][B*S][D]
       const __bf16* __restrict__ Kb,   // [H][B*S][D]
       const __bf16* __restrict__ VbT,  // [H][D][B*S]  (transposed)
       __bf16* __restrict__ Zcat,       // [B*S][H*D]
       float*  __restrict__ attnMean) { // [B][S][S]
  extern __shared__ char smem[];
  __bf16* sP    = (__bf16*)smem;                    // [16][2048]
  float*  sMean = (float*)(smem + 16 * 2048 * 2);   // [16][2048]

  const int tid  = threadIdx.x;
  const int lane = tid & 31;
  const int wave = tid >> 5;
  const int b    = blockIdx.y;
  const long s0  = (long)blockIdx.x * 16;
  const int ml   = lane & 15;
  const int kh   = (lane < 16) ? 0 : 8;
  const int rh   = (lane < 16) ? 0 : 8;

  for (int i = tid; i < 16 * 2048; i += 256) sMean[i] = 0.0f;
  __syncthreads();

  for (int h = 0; h < H_; ++h) {
    const __bf16* Qh = Qb + ((long)h * M_ + (long)b * S_ + s0) * D_;
    const __bf16* Kh = Kb + ((long)h * M_ + (long)b * S_) * D_;

    // ---- stage 1: scores for t in [wave*256, wave*256+256) ----
    const int tbase = wave * 256;
    const __bf16* qrow = Qh + (long)ml * D_ + kh;
    bf16x16 qf[8];
#pragma unroll
    for (int e = 0; e < 8; ++e) qf[e] = frag_ld_g(qrow + e * 32);

    for (int j = 0; j < 16; ++j) {
      const int t0 = tbase + j * 16;
      const __bf16* krow = Kh + (long)(t0 + ml) * D_ + kh;
      f32x8 acc = {};
#pragma unroll
      for (int e = 0; e < 8; ++e) {
        bf16x16 bfr = frag_ld_g(krow + e * 32);
        acc = wmma_bf16(qf[e], bfr, acc);
      }
#pragma unroll
      for (int i = 0; i < 8; ++i)
        sP[(rh + i) * 2048 + t0 + ml] = (__bf16)acc[i];
    }
    __syncthreads();

    // ---- stage 2: softmax rows [2*wave, 2*wave+1] ----
    for (int rr = 0; rr < 2; ++rr) {
      const int r = wave * 2 + rr;
      __bf16* row = sP + r * 2048;
      float mx = -3.0e38f;
      for (int c = lane; c < 2048; c += 32) mx = fmaxf(mx, (float)row[c]);
      for (int o = 16; o; o >>= 1) mx = fmaxf(mx, __shfl_xor(mx, o, 32));
      float sum = 0.0f;
      for (int c = lane; c < 2048; c += 32) {
        float p = __expf((float)row[c] - mx);
        sum += p;
        row[c] = (__bf16)p;
      }
      for (int o = 16; o; o >>= 1) sum += __shfl_xor(sum, o, 32);
      const float inv = 1.0f / sum;
      float* mrow = sMean + r * 2048;
      for (int c = lane; c < 2048; c += 32) {
        float p = (float)row[c] * inv;
        row[c] = (__bf16)p;
        mrow[c] += p;
      }
    }
    __syncthreads();

    // ---- stage 3: zh slab, e in [wave*32, wave*32+32) ----
    const int e0 = wave * 32;
    const __bf16* vbase = VbT + (long)h * D_ * M_ + (long)b * S_;
    f32x8 acc0 = {}, acc1 = {};
    for (int t0 = 0; t0 < S_; t0 += 32) {
      bf16x16 afr = frag_ld_g(sP + (long)ml * 2048 + t0 + kh);
      bf16x16 b0  = frag_ld_g(vbase + (long)(e0 + ml) * M_ + t0 + kh);
      bf16x16 b1  = frag_ld_g(vbase + (long)(e0 + 16 + ml) * M_ + t0 + kh);
      acc0 = wmma_bf16(afr, b0, acc0);
      acc1 = wmma_bf16(afr, b1, acc1);
    }
#pragma unroll
    for (int i = 0; i < 8; ++i) {
      long rowm = (long)b * S_ + s0 + rh + i;
      Zcat[rowm * HD_ + (h * D_ + e0 + ml)]      = (__bf16)acc0[i];
      Zcat[rowm * HD_ + (h * D_ + e0 + 16 + ml)] = (__bf16)acc1[i];
    }
    __syncthreads();   // protect sP before next head
  }

  // ---- write attn_mean = sum/H once ----
  for (int i = tid * 4; i < 16 * 2048; i += 256 * 4) {
    int r = i >> 11, c = i & 2047;
    float4 v = *(const float4*)(sMean + r * 2048 + c);
    v.x *= 0.125f; v.y *= 0.125f; v.z *= 0.125f; v.w *= 0.125f;
    *(float4*)(attnMean + ((long)b * S_ + s0 + r) * S_ + c) = v;
  }
}

// ---------------------------------------------------------------------------
// LayerNorm over D=256 (one row per 256-thread block, 8 wave32 partials)
// ---------------------------------------------------------------------------
__global__ void __launch_bounds__(256)
ln_k(const float* __restrict__ in, const float* __restrict__ res,
     const float* __restrict__ g, const float* __restrict__ be,
     float* __restrict__ outF, __bf16* __restrict__ outB) {
  __shared__ float red[8];
  const long row = blockIdx.x;
  const int  t   = threadIdx.x;
  float x = in[row * D_ + t];
  if (res) x += res[row * D_ + t];

  float v = x;
  for (int o = 16; o; o >>= 1) v += __shfl_xor(v, o, 32);
  if ((t & 31) == 0) red[t >> 5] = v;
  __syncthreads();
  float tot = 0.0f;
#pragma unroll
  for (int i = 0; i < 8; ++i) tot += red[i];
  const float mean = tot * (1.0f / D_);
  __syncthreads();

  const float d = x - mean;
  float v2 = d * d;
  for (int o = 16; o; o >>= 1) v2 += __shfl_xor(v2, o, 32);
  if ((t & 31) == 0) red[t >> 5] = v2;
  __syncthreads();
  float tot2 = 0.0f;
#pragma unroll
  for (int i = 0; i < 8; ++i) tot2 += red[i];
  const float var = tot2 * (1.0f / D_);

  const float y = d * rsqrtf(var + 1e-5f) * g[t] + be[t];
  if (outF) outF[row * D_ + t] = y;
  if (outB) outB[row * D_ + t] = (__bf16)y;
}

// ---------------------------------------------------------------------------
// Host-side orchestration (all launches on `stream`, graph-capture safe)
// ---------------------------------------------------------------------------
extern "C" void kernel_launch(void* const* d_in, const int* in_sizes, int n_in,
                              void* d_out, int out_size, void* d_ws, size_t ws_size,
                              hipStream_t stream) {
  const float* X  = (const float*)d_in[0];
  const float* Wq = (const float*)d_in[1];
  const float* Wk = (const float*)d_in[2];
  const float* Wv = (const float*)d_in[3];
  const float* Wz = (const float*)d_in[4];
  const float* bz = (const float*)d_in[5];
  const float* W1 = (const float*)d_in[6];
  const float* b1 = (const float*)d_in[7];
  const float* W2 = (const float*)d_in[8];
  const float* b2 = (const float*)d_in[9];
  const float* g1  = (const float*)d_in[10];
  const float* be1 = (const float*)d_in[11];
  const float* g2  = (const float*)d_in[12];
  const float* be2 = (const float*)d_in[13];

  float* outZ    = (float*)d_out;            // [B,S,D]
  float* outAttn = outZ + M_ * D_;           // [B,S,S]

  // ---- workspace carve-up ----
  char* ws = (char*)d_ws;
  size_t off = 0;
  auto alloc = [&](size_t bytes) { char* p = ws + off; off = (off + bytes + 255) & ~(size_t)255; return p; };
  __bf16* Xb   = (__bf16*)alloc(M_ * D_ * 2);
  __bf16* WqT  = (__bf16*)alloc((size_t)H_ * D_ * D_ * 2);
  __bf16* WkT  = (__bf16*)alloc((size_t)H_ * D_ * D_ * 2);
  __bf16* WvT  = (__bf16*)alloc((size_t)H_ * D_ * D_ * 2);
  __bf16* WzT  = (__bf16*)alloc((size_t)HD_ * D_ * 2);
  __bf16* W1T  = (__bf16*)alloc((size_t)FD_ * D_ * 2);
  __bf16* W2T  = (__bf16*)alloc((size_t)D_ * FD_ * 2);
  __bf16* Qb   = (__bf16*)alloc((size_t)H_ * M_ * D_ * 2);
  __bf16* Kb   = (__bf16*)alloc((size_t)H_ * M_ * D_ * 2);
  __bf16* VbT  = (__bf16*)alloc((size_t)H_ * D_ * M_ * 2);
  __bf16* Zcat = (__bf16*)alloc((size_t)M_ * HD_ * 2);
  float*  zres  = (float*)alloc((size_t)M_ * D_ * 4);
  float*  zln1f = (float*)alloc((size_t)M_ * D_ * 4);
  __bf16* zln1b = (__bf16*)alloc((size_t)M_ * D_ * 2);
  __bf16* Hff   = (__bf16*)alloc((size_t)M_ * FD_ * 2);
  float*  zff   = (float*)alloc((size_t)M_ * D_ * 4);
  (void)ws_size; (void)in_sizes; (void)n_in; (void)out_size;

  // ---- 1. casts / weight transposes ----
  cast_bf16_k<<<1024, 256, 0, stream>>>(X, Xb, M_ * D_);
  dim3 tb(16, 16);
  transpose_cast_k<<<dim3(D_/16, D_/16, H_), tb, 0, stream>>>(Wq, WqT, D_, D_, (long)D_*D_, (long)D_*D_);
  transpose_cast_k<<<dim3(D_/16, D_/16, H_), tb, 0, stream>>>(Wk, WkT, D_, D_, (long)D_*D_, (long)D_*D_);
  transpose_cast_k<<<dim3(D_/16, D_/16, H_), tb, 0, stream>>>(Wv, WvT, D_, D_, (long)D_*D_, (long)D_*D_);
  transpose_cast_k<<<dim3(D_/16, HD_/16, 1), tb, 0, stream>>>(Wz, WzT, (int)HD_, D_, 0, 0);
  transpose_cast_k<<<dim3(FD_/16, D_/16, 1), tb, 0, stream>>>(W1, W1T, D_, FD_, 0, 0);
  transpose_cast_k<<<dim3(D_/16, FD_/16, 1), tb, 0, stream>>>(W2, W2T, FD_, D_, 0, 0);

  const float scale = 0.25f;  // D^-0.25 = 256^-0.25
  // ---- 2. QKV projections (bf16 WMMA), per-head via grid.z ----
  gemm_bf16_wmma<<<dim3(M_/128, D_/64, H_), 256, 0, stream>>>(
      Xb, D_, 0, WqT, D_, (long)D_*D_, nullptr, nullptr, 0, scale, 0,
      nullptr, Qb, D_, (long)M_*D_, nullptr, 0, 0, D_);
  gemm_bf16_wmma<<<dim3(M_/128, D_/64, H_), 256, 0, stream>>>(
      Xb, D_, 0, WkT, D_, (long)D_*D_, nullptr, nullptr, 0, scale, 0,
      nullptr, Kb, D_, (long)M_*D_, nullptr, 0, 0, D_);
  gemm_bf16_wmma<<<dim3(M_/128, D_/64, H_), 256, 0, stream>>>(
      Xb, D_, 0, WvT, D_, (long)D_*D_, nullptr, nullptr, 0, 1.0f, 0,
      nullptr, nullptr, 0, 0, VbT, M_, (long)D_*M_, D_);   // V stored transposed

  // ---- 3. attention (scores + softmax + P@V + attn_mean) ----
  const size_t attnLds = 16 * 2048 * 2 + 16 * 2048 * 4;    // 192 KB
  attn_k<<<dim3(S_/16, B_), 256, attnLds, stream>>>(Qb, Kb, VbT, Zcat, outAttn);

  // ---- 4. output projection + bias + residual (fp32 out) ----
  gemm_bf16_wmma<<<dim3(M_/128, D_/64, 1), 256, 0, stream>>>(
      Zcat, HD_, 0, WzT, HD_, 0, bz, X, D_, 1.0f, 0,
      zres, nullptr, D_, 0, nullptr, 0, 0, (int)HD_);

  // ---- 5. LayerNorm 1 ----
  ln_k<<<M_, D_, 0, stream>>>(zres, nullptr, g1, be1, zln1f, zln1b);

  // ---- 6. MLP ----
  gemm_bf16_wmma<<<dim3(M_/128, FD_/64, 1), 256, 0, stream>>>(
      zln1b, D_, 0, W1T, D_, 0, b1, nullptr, 0, 1.0f, 1,
      nullptr, Hff, FD_, 0, nullptr, 0, 0, D_);
  gemm_bf16_wmma<<<dim3(M_/128, D_/64, 1), 256, 0, stream>>>(
      Hff, FD_, 0, W2T, FD_, 0, b2, nullptr, 0, 1.0f, 0,
      zff, nullptr, D_, 0, nullptr, 0, 0, FD_);

  // ---- 7. LayerNorm 2 (+residual) -> final z output ----
  ln_k<<<M_, D_, 0, stream>>>(zff, zln1f, g2, be2, outZ, nullptr);
}
